// TGNN_18124761989843
// MI455X (gfx1250) — matrix operand order repaired and checked by reading, not compiled, not run on hardware
//
#include <hip/hip_runtime.h>
#include <math.h>

// ---------------- types for CDNA5 WMMA ----------------
typedef __bf16 bf16_t;
typedef __attribute__((ext_vector_type(16))) __bf16 v16bf;
typedef __attribute__((ext_vector_type(8)))  __bf16 v8bf;
typedef __attribute__((ext_vector_type(8)))  float  v8f;

#define N_NODES 512
#define T_LEN   128
#define N_E     4096
#define C0      64
#define EMB     256
#define C1      128
#define C2      128
#define NGRAPH  8
#define NPG     64           // nodes per graph
#define EPG     (N_E/NGRAPH) // edges per graph (contiguous slabs)
#define EPS_BN  1e-5f

// =====================================================================
// bf16 WMMA GEMM:  C[M,N] (f32) = A[M,K] (bf16) @ B[K,N] (bf16)
//  - B passed TRANSPOSED: Bt[N][K] so each lane's B fragment is a
//    contiguous 32B global run (2x global_load_b128, no LDS, no shuffles).
//  - A staged in LDS in WMMA *fragment order*: writers emit one 16B
//    ds_store_b128 per 8 elements; readers pull v16bf operands directly.
//  - A-row addressing supports overlapping conv windows:
//      elem(m,k) = A[(m/rows_per_grp)*grp_stride + (m%rows_per_grp)*row_stride + k]
// Block tile 128x128, K-step 32; 8 waves as 4(M)x2(N); wave = 32x64
// => 8 v_wmma per wave per K-step. Requires N%128==0, K%32==0.
// =====================================================================
#define BM 128
#define BN 128
#define BK 32

__global__ __launch_bounds__(256) void k_gemm_bf16(
    const bf16_t* __restrict__ A, const bf16_t* __restrict__ Bt,
    float* __restrict__ C, int M, int N, int K,
    int rows_per_grp, long long grp_stride, int row_stride) {
  // A tile, fragment-ordered: [row-tile 0..7][lane 0..31] -> v16bf operand
  __shared__ v16bf sAf[BM / 16][32];

  const int tid  = threadIdx.x;
  const int lane = tid & 31;
  const int wave = tid >> 5;
  const int wm   = wave & 3;   // M slot (32 rows each)
  const int wn   = wave >> 2;  // N slot (64 cols each)
  const int mblk = blockIdx.y * BM;
  const int nblk = blockIdx.x * BN;
  const int lh   = lane >> 4;
  const int l15  = lane & 15;
  const bool fullM = (mblk + BM) <= M;

  // ---- writer precompute: each thread owns row wr, 16 k's (two 8-chunks) ----
  const int wr  = tid >> 1;            // row in tile 0..127
  const int wk0 = (tid & 1) * 16;      // k base within tile: 0 or 16
  const int wmg = mblk + wr;
  const bool wvalid = wmg < M;
  size_t aoff0 = 0;
  if (wvalid)
    aoff0 = (size_t)(wmg / rows_per_grp) * (size_t)grp_stride
          + (size_t)(wmg % rows_per_grp) * (size_t)row_stride;
  bf16_t* sflat = (bf16_t*)sAf;
  // LDS chunk index for (wr, kk): lane' = (wr&15)+16*(kk>>4), e0 = ((kk>>3)&1)*8
  int ldst[2];
  #pragma unroll
  for (int c2 = 0; c2 < 2; ++c2) {
    int kk = wk0 + 8 * c2;
    int lp = (wr & 15) + 16 * (kk >> 4);
    int e0 = ((kk >> 3) & 1) * 8;
    ldst[c2] = ((wr >> 4) * 32 + lp) * 16 + e0;
  }

  // ---- B fragment base pointers (transposed weights, L2-resident) ----
  const bf16_t* bp[4];
  #pragma unroll
  for (int j = 0; j < 4; ++j) {
    int n = nblk + wn * 64 + j * 16 + l15;
    bp[j] = Bt + (size_t)n * (size_t)K + (size_t)(lh * 16);
  }

  v8f acc[2][4] = {};

  for (int k0 = 0; k0 < K; k0 += BK) {
    // stage A tile (fragment order): one 16B LDS store per 8 elements
    #pragma unroll
    for (int c2 = 0; c2 < 2; ++c2) {
      int kk = wk0 + 8 * c2;
      v8bf v;
      #pragma unroll
      for (int e = 0; e < 8; ++e) v[e] = (bf16_t)0.0f;
      if (wvalid) v = *(const v8bf*)(A + aoff0 + (size_t)(k0 + kk));
      *(v8bf*)(sflat + ldst[c2]) = v;
    }
    __syncthreads();

    v16bf a0 = sAf[wm * 2 + 0][lane];
    v16bf a1 = sAf[wm * 2 + 1][lane];
    #pragma unroll
    for (int j = 0; j < 4; ++j) {
      v16bf b = *(const v16bf*)(bp[j] + k0);
      acc[0][j] = __builtin_amdgcn_wmma_f32_16x16x32_bf16(false, a0, false, b, (short)0, acc[0][j], false, false);
      acc[1][j] = __builtin_amdgcn_wmma_f32_16x16x32_bf16(false, a1, false, b, (short)0, acc[1][j], false, false);
    }
    __syncthreads();
  }

  // ---- store D (f32 16x16 layout: VGPR r -> row r+8*lh, lane -> col) ----
  #pragma unroll
  for (int i = 0; i < 2; ++i) {
    #pragma unroll
    for (int j = 0; j < 4; ++j) {
      int n     = nblk + wn * 64 + j * 16 + l15;
      int mbase = mblk + wm * 32 + i * 16 + 8 * lh;
      float* cp = C + (size_t)mbase * (size_t)N + (size_t)n;
      if (fullM) {
        #pragma unroll
        for (int r = 0; r < 8; ++r) cp[(size_t)r * N] = acc[i][j][r];
      } else {
        #pragma unroll
        for (int r = 0; r < 8; ++r)
          if (mbase + r < M) cp[(size_t)r * N] = acc[i][j][r];
      }
    }
  }
}

// ---------------- f32 -> bf16 (plain) ----------------
__global__ void k_cvt_bf16(const float* __restrict__ in, bf16_t* __restrict__ out, int n) {
  int i = blockIdx.x * blockDim.x + threadIdx.x;
  if (i < n) out[i] = (bf16_t)in[i];
}

// ---------------- f32 (K x N) -> bf16 transposed (N x K) ----------------
__global__ void k_cvt_bf16_t(const float* __restrict__ W, bf16_t* __restrict__ Bt,
                             int K, int N) {
  int i = blockIdx.x * blockDim.x + threadIdx.x;
  if (i >= K * N) return;
  int k = i / N, n = i % N;
  Bt[(size_t)n * K + k] = (bf16_t)W[i];
}

// ---------------- conv1: (512,128,1) -> (512,128,64), k=7 SAME ----------------
__global__ void k_conv1(const float* __restrict__ x, const float* __restrict__ W1,
                        float* __restrict__ y) {
  int i = blockIdx.x * blockDim.x + threadIdx.x;
  if (i >= N_NODES * T_LEN * C0) return;
  int c = i & 63, t = (i >> 6) & 127, n = i >> 13;
  float acc = 0.f;
  #pragma unroll
  for (int k = 0; k < 7; ++k) {
    int tt = t + k - 3;
    if (tt >= 0 && tt < T_LEN) acc += x[n * T_LEN + tt] * W1[k * C0 + c];
  }
  y[i] = acc;
}

// ---------------- training-mode BN stats ----------------
__global__ void k_stats(const float* __restrict__ y, const float* __restrict__ gamma,
                        const float* __restrict__ beta, float* __restrict__ st,
                        int C, int S) {
  int c = blockIdx.x, tid = threadIdx.x;
  float s = 0.f, s2 = 0.f;
  for (int i = tid; i < S; i += 256) {
    float v = y[(size_t)i * C + c];
    s += v; s2 += v * v;
  }
  __shared__ float sh[256], sh2[256];
  sh[tid] = s; sh2[tid] = s2; __syncthreads();
  for (int w = 128; w > 0; w >>= 1) {
    if (tid < w) { sh[tid] += sh[tid + w]; sh2[tid] += sh2[tid + w]; }
    __syncthreads();
  }
  if (tid == 0) {
    float mu  = sh[0] / (float)S;
    float var = sh2[0] / (float)S - mu * mu;
    st[c]         = mu;
    st[C + c]     = gamma[c] * rsqrtf(var + EPS_BN);
    st[2 * C + c] = beta[c];
  }
}

// ---------------- position-encoder MLP ----------------
__global__ void k_pe(const float* __restrict__ pos, const float* __restrict__ Wp1,
                     const float* __restrict__ bp1, const float* __restrict__ Wp2,
                     const float* __restrict__ bp2, float* __restrict__ pe) {
  int i = blockIdx.x * blockDim.x + threadIdx.x;
  if (i >= N_NODES * C0) return;
  int c = i & 63, n = i >> 6;
  float p0 = pos[n * 3 + 0], p1 = pos[n * 3 + 1], p2 = pos[n * 3 + 2];
  float acc = bp2[c];
  for (int j = 0; j < C0; ++j) {
    float t = p0 * Wp1[j] + p1 * Wp1[C0 + j] + p2 * Wp1[2 * C0 + j] + bp1[j];
    t = fmaxf(t, 0.f);
    acc += t * Wp2[j * C0 + c];
  }
  pe[i] = acc;
}

// ---------------- BN + ReLU + pos-enc add, emit bf16 h0 ----------------
__global__ void k_bn_relu_pe(const float* __restrict__ y, const float* __restrict__ st,
                             const float* __restrict__ pe, bf16_t* __restrict__ H) {
  int i = blockIdx.x * blockDim.x + threadIdx.x;
  if (i >= N_NODES * T_LEN * C0) return;
  int c = i & 63, n = i >> 13;
  float v = st[C0 + c] * (y[i] - st[c]) + st[2 * C0 + c];
  H[i] = (bf16_t)(fmaxf(v, 0.f) + pe[(n << 6) + c]);
}

// ---------------- symmetric-norm inverse sqrt degrees ----------------
__global__ void k_dinv(const int* __restrict__ dst, float* __restrict__ dinv) {
  int n = blockIdx.x * blockDim.x + threadIdx.x;
  if (n >= N_NODES) return;
  int g = n / NPG, cnt = 0;
  for (int e = g * EPG; e < (g + 1) * EPG; ++e) cnt += (dst[e] == n) ? 1 : 0;
  dinv[n] = rsqrtf((float)cnt + 1.0f);
}

// ---------------- GCN aggregation (deterministic gather) -----------
// hW: (node, t, f) f32; writes H (bf16) = relu(agg + self + bias). F power of 2.
__global__ __launch_bounds__(256) void k_gcn_agg(
    const float* __restrict__ hW, const float* __restrict__ dinv,
    const int* __restrict__ src, const int* __restrict__ dst,
    const float* __restrict__ bias, bf16_t* __restrict__ Hout, int F) {
  int node = blockIdx.x, tid = threadIdx.x;
  __shared__ int   sCnt;
  __shared__ int   sSrc[EPG];
  __shared__ float sNorm[EPG];
  float dn = dinv[node];
  if (tid == 0) {
    int g = node / NPG, c = 0;
    for (int e = g * EPG; e < (g + 1) * EPG; ++e) {
      if (dst[e] == node) { sSrc[c] = src[e]; sNorm[c] = dinv[src[e]] * dn; ++c; }
    }
    sCnt = c;
  }
  __syncthreads();
  int cnt = sCnt;
  float selfs = dn * dn;
  size_t base = (size_t)node * T_LEN * F;
  int tot = T_LEN * F, fm = F - 1;
  for (int i = tid; i < tot; i += 256) {
    float acc = selfs * hW[base + i];
    for (int j = 0; j < cnt; ++j)
      acc += sNorm[j] * hW[(size_t)sSrc[j] * T_LEN * F + i];
    acc += bias[i & fm];
    Hout[base + i] = (bf16_t)fmaxf(acc, 0.f);
  }
}

// ---------------- BN apply + avgpool(4) + ReLU ----------------
__global__ void k_bn_pool_relu(const float* __restrict__ y, const float* __restrict__ st,
                               float* __restrict__ out, int NN, int TP, int Tin, int C) {
  int i = blockIdx.x * blockDim.x + threadIdx.x;
  if (i >= NN * TP * C) return;
  int c = i % C, tp = (i / C) % TP, n = i / (C * TP);
  float mu = st[c], sc = st[C + c], bt = st[2 * C + c];
  float acc = 0.f;
  #pragma unroll
  for (int p = 0; p < 4; ++p)
    acc += sc * (y[((size_t)n * Tin + tp * 4 + p) * C + c] - mu) + bt;
  out[i] = fmaxf(acc * 0.25f, 0.f);
}

// ---------------- per-graph mean over 64 nodes, emit bf16 ----------------
__global__ void k_graphmean(const float* __restrict__ h3, bf16_t* __restrict__ gb) {
  int i = blockIdx.x * blockDim.x + threadIdx.x;
  if (i >= NGRAPH * 30 * C1) return;
  int c = i & 127, tp = (i >> 7) % 30, b = i / (128 * 30);
  float s = 0.f;
  for (int nn = 0; nn < NPG; ++nn)
    s += h3[(((size_t)(b * NPG + nn)) * 30 + tp) * C1 + c];
  gb[i] = (bf16_t)(s * (1.0f / NPG));
}

// ---------------- classifier head + log_softmax ----------------
__global__ void k_head(const float* __restrict__ h4, const float* __restrict__ Wd,
                       const float* __restrict__ bd, float* __restrict__ out) {
  int b = blockIdx.x, tid = threadIdx.x;  // 128 threads
  float l[4] = {0.f, 0.f, 0.f, 0.f};
  for (int i = tid; i < 7 * C2; i += 128) {
    float v = h4[b * 7 * C2 + i];
    #pragma unroll
    for (int j = 0; j < 4; ++j) l[j] += v * Wd[i * 4 + j];
  }
  __shared__ float sh[4][128];
  #pragma unroll
  for (int j = 0; j < 4; ++j) sh[j][tid] = l[j];
  __syncthreads();
  for (int w = 64; w > 0; w >>= 1) {
    if (tid < w) {
      #pragma unroll
      for (int j = 0; j < 4; ++j) sh[j][tid] += sh[j][tid + w];
    }
    __syncthreads();
  }
  if (tid == 0) {
    float lg[4], mx = -1e30f;
    #pragma unroll
    for (int j = 0; j < 4; ++j) { lg[j] = sh[j][0] + bd[j]; mx = fmaxf(mx, lg[j]); }
    float se = 0.f;
    #pragma unroll
    for (int j = 0; j < 4; ++j) se += expf(lg[j] - mx);
    float lse = mx + logf(se);
    #pragma unroll
    for (int j = 0; j < 4; ++j) out[b * 4 + j] = lg[j] - lse;
  }
}

// =====================================================================
extern "C" void kernel_launch(void* const* d_in, const int* in_sizes, int n_in,
                              void* d_out, int out_size, void* d_ws, size_t ws_size,
                              hipStream_t stream) {
  (void)in_sizes; (void)n_in; (void)out_size; (void)ws_size;
  const float* x   = (const float*)d_in[0];
  const float* pos = (const float*)d_in[1];
  const int*   ei  = (const int*)  d_in[2];   // [src(4096); dst(4096)]
  const float* W1  = (const float*)d_in[4];
  const float* g1  = (const float*)d_in[5];
  const float* b1  = (const float*)d_in[6];
  const float* Wp1 = (const float*)d_in[7];
  const float* bp1 = (const float*)d_in[8];
  const float* Wp2 = (const float*)d_in[9];
  const float* bp2 = (const float*)d_in[10];
  const float* Wg1 = (const float*)d_in[11];
  const float* bg1 = (const float*)d_in[12];
  const float* Wg2 = (const float*)d_in[13];
  const float* bg2 = (const float*)d_in[14];
  const float* W2  = (const float*)d_in[15];
  const float* g2  = (const float*)d_in[16];
  const float* b2  = (const float*)d_in[17];
  const float* W3  = (const float*)d_in[18];
  const float* g3  = (const float*)d_in[19];
  const float* b3  = (const float*)d_in[20];
  const float* Wd  = (const float*)d_in[21];
  const float* bd  = (const float*)d_in[22];
  float* out = (float*)d_out;
  const int* srcA = ei;
  const int* dstA = ei + N_E;

  // ---- workspace carve-up ----
  char* w = (char*)d_ws;
  size_t off = 0;
  auto take = [&](size_t bytes) -> char* {
    char* p = w + off; off += (bytes + 255) & ~(size_t)255; return p;
  };
  bf16_t* H    = (bf16_t*)take((size_t)N_NODES * T_LEN * EMB * 2); // activations (bf16)
  float*  HW   = (float*) take((size_t)N_NODES * T_LEN * EMB * 4); // GEMM outputs (f32)
  float*  pe   = (float*) take((size_t)N_NODES * C0 * 4);
  float*  dinv = (float*) take((size_t)N_NODES * 4);
  float*  st   = (float*) take(3 * 128 * 4);
  bf16_t* Wg1t = (bf16_t*)take((size_t)C0 * EMB * 2);   // EMB x C0
  bf16_t* Wg2t = (bf16_t*)take((size_t)EMB * EMB * 2);  // EMB x EMB
  bf16_t* W2t  = (bf16_t*)take((size_t)7 * EMB * C1 * 2); // C1 x 1792
  bf16_t* W3t  = (bf16_t*)take((size_t)3 * C1 * C2 * 2);  // C2 x 384
  float*  h3   = (float*) take((size_t)N_NODES * 30 * C1 * 4);
  bf16_t* gb   = (bf16_t*)take((size_t)NGRAPH * 30 * C1 * 2);
  float*  y3   = (float*) take((size_t)NGRAPH * 28 * C2 * 4);
  float*  h4   = (float*) take((size_t)NGRAPH * 7 * C2 * 4);

  // ---- transposed bf16 weight conversions ----
  auto cvt_t = [&](const float* s, bf16_t* d, int K, int N) {
    int n = K * N;
    k_cvt_bf16_t<<<(n + 255) / 256, 256, 0, stream>>>(s, d, K, N);
  };
  cvt_t(Wg1, Wg1t, C0, EMB);
  cvt_t(Wg2, Wg2t, EMB, EMB);
  cvt_t(W2,  W2t,  7 * EMB, C1);
  cvt_t(W3,  W3t,  3 * C1, C2);

  // ---- stage 1: conv1 + BN + ReLU + pos-encoder ----
  int n1 = N_NODES * T_LEN * C0;
  k_conv1<<<(n1 + 255) / 256, 256, 0, stream>>>(x, W1, HW);
  k_stats<<<C0, 256, 0, stream>>>(HW, g1, b1, st, C0, N_NODES * T_LEN);
  k_pe<<<(N_NODES * C0 + 255) / 256, 256, 0, stream>>>(pos, Wp1, bp1, Wp2, bp2, pe);
  k_bn_relu_pe<<<(n1 + 255) / 256, 256, 0, stream>>>(HW, st, pe, H);
  k_dinv<<<(N_NODES + 255) / 256, 256, 0, stream>>>(dstA, dinv);

  // ---- GCN layer 1: (65536 x 64) @ (64 x 256), aggregate ----
  {
    int M = N_NODES * T_LEN, K = C0, N = EMB;
    dim3 grid(N / BN, (M + BM - 1) / BM);
    k_gemm_bf16<<<grid, 256, 0, stream>>>(H, Wg1t, HW, M, N, K, M, 0LL, K);
    k_gcn_agg<<<N_NODES, 256, 0, stream>>>(HW, dinv, srcA, dstA, bg1, H, EMB);
  }
  // ---- GCN layer 2: (65536 x 256) @ (256 x 256), aggregate ----
  {
    int M = N_NODES * T_LEN, K = EMB, N = EMB;
    dim3 grid(N / BN, (M + BM - 1) / BM);
    k_gemm_bf16<<<grid, 256, 0, stream>>>(H, Wg2t, HW, M, N, K, M, 0LL, K);
    k_gcn_agg<<<N_NODES, 256, 0, stream>>>(HW, dinv, srcA, dstA, bg2, H, EMB);
  }

  // ---- conv2 (valid,k=7) as implicit-im2col GEMM: M=62464, K=1792, N=128 ----
  {
    int M = N_NODES * 122, K = 7 * EMB, N = C1;
    dim3 grid(N / BN, (M + BM - 1) / BM);
    k_gemm_bf16<<<grid, 256, 0, stream>>>(H, W2t, HW, M, N, K,
                                          122, (long long)T_LEN * EMB, EMB);
    k_stats<<<C1, 256, 0, stream>>>(HW, g2, b2, st, C1, M);
    int n2 = N_NODES * 30 * C1;
    k_bn_pool_relu<<<(n2 + 255) / 256, 256, 0, stream>>>(HW, st, h3, N_NODES, 30, 122, C1);
  }

  // ---- global mean pool per graph ----
  {
    int n3 = NGRAPH * 30 * C1;
    k_graphmean<<<(n3 + 255) / 256, 256, 0, stream>>>(h3, gb);
  }

  // ---- conv3 (valid,k=3) as GEMM: M=224, K=384, N=128 ----
  {
    int M = NGRAPH * 28, K = 3 * C1, N = C2;
    dim3 grid(N / BN, (M + BM - 1) / BM);
    k_gemm_bf16<<<grid, 256, 0, stream>>>(gb, W3t, y3, M, N, K,
                                          28, (long long)30 * C1, C1);
    k_stats<<<C2, 256, 0, stream>>>(y3, g3, b3, st, C2, M);
    int n4 = NGRAPH * 7 * C2;
    k_bn_pool_relu<<<(n4 + 255) / 256, 256, 0, stream>>>(y3, st, h4, NGRAPH, 7, 28, C2);
  }

  // ---- classifier + log_softmax ----
  k_head<<<NGRAPH, 128, 0, stream>>>(h4, Wd, bd, out);
}
